// MolGINE_37486474559539
// MI455X (gfx1250) — compile-verified, matrix operand order checked by reading, not run on hardware
//
#include <hip/hip_runtime.h>
#include <hip/hip_bf16.h>
#include <math.h>

// ---------------------------------------------------------------------------
// MolGINE forward for MI455X (gfx1250, wave32, WMMA).
// Dense GEMMs: v_wmma_f32_16x16x32_f16, f32 accumulate, K-tile 64,
// LDS double-buffered, software-pipelined global->reg->LDS staging.
// ---------------------------------------------------------------------------

typedef _Float16 v16h __attribute__((ext_vector_type(16)));
typedef _Float16 v8h  __attribute__((ext_vector_type(8)));
typedef float    v8f  __attribute__((ext_vector_type(8)));

#define BN_INV 0.9999950000374997f   // 1/sqrt(1+1e-5), eval-mode BN

static constexpr int NN = 100000;   // nodes
static constexpr int NE = 400000;   // edges
static constexpr int NG = 4096;     // graphs
static constexpr int HH = 256;      // hidden
static constexpr int LL = 5;        // layers
static constexpr int JK = HH * (LL + 1);   // 1536
static constexpr int GF = 10;
static constexpr int PCW = JK + GF;        // 1546

static constexpr int ASTR = 72;     // LDS row stride in f16 (144B, 16B mult)

// ---------------------------------------------------------------------------
// Weight prep: W[K][Nc] f32 -> Wt[Nc][Kpad] f16 (transposed, zero K-padded).
// ---------------------------------------------------------------------------
__global__ void prep_wt_kernel(const float* __restrict__ W, int K, int Nc,
                               int Kpad, _Float16* __restrict__ Wt) {
    int t = blockIdx.x * blockDim.x + threadIdx.x;
    if (t >= Nc * Kpad) return;
    int n = t / Kpad, k = t - n * Kpad;
    Wt[(size_t)n * Kpad + k] = (k < K) ? (_Float16)W[(size_t)k * Nc + n]
                                       : (_Float16)0.0f;
}

// ---------------------------------------------------------------------------
// A-tile staging helpers: 32 k-values per thread (BK=64, 2 threads/row).
// Fast path: 16x float2 (lda may be only 8B aligned, e.g. lda=38).
// Tail path only taken in the zero-padded K region (Wp GEMM).
// ---------------------------------------------------------------------------
__device__ __forceinline__ void stage_a32(const float* __restrict__ A, int lda,
                                          int rowc, int kact, int kbase,
                                          float (&af)[32]) {
    const float* ap = A + (size_t)rowc * lda + kbase;
    if (kbase + 32 <= kact) {
#pragma unroll
        for (int i = 0; i < 16; ++i) {
            float2 v = *(const float2*)(ap + 2 * i);
            af[2 * i]     = v.x;
            af[2 * i + 1] = v.y;
        }
    } else {
#pragma unroll
        for (int i = 0; i < 32; ++i)
            af[i] = (kbase + i < kact) ? ap[i] : 0.0f;
    }
}

__device__ __forceinline__ void commit_a32(_Float16* __restrict__ dst,
                                           const float (&af)[32]) {
#pragma unroll
    for (int q = 0; q < 4; ++q) {
        v8h p;
#pragma unroll
        for (int i = 0; i < 8; ++i) p[i] = (_Float16)af[q * 8 + i];
        *(v8h*)(dst + q * 8) = p;
    }
}

// ---------------------------------------------------------------------------
// WMMA GEMM: C[M,Ntot] = epilogue(A[M,Kact] * Bt[Ntot,Kpad]^T + bias)
//   Bt pre-transposed f16, row stride Kpad (multiple of 64).
//   mode 0: none | 1: relu | 2: relu(g*(x*BN_INV)+bb) (+res if non-null)
// Block tile 128(M) x 64(N) x 64(K), 8 wave32 waves, LDS ping-pong,
// 1 barrier per K-step -> 8 WMMAs per wave per barrier.
// ---------------------------------------------------------------------------
__global__ __launch_bounds__(256)
void wmma_gemm_kernel(const float* __restrict__ A, int lda,
                      const _Float16* __restrict__ Bt, int kpad,
                      float* __restrict__ C, int ldc,
                      const float* __restrict__ bias,
                      const float* __restrict__ g, const float* __restrict__ bb,
                      const float* __restrict__ res, int ldres,
                      int M, int kact, int mode) {
    __shared__ _Float16 As[2][128 * ASTR];
    __shared__ _Float16 Bs[2][64 * ASTR];

    const int tid  = threadIdx.x;
    const int lane = tid & 31;
    const int w    = tid >> 5;
    const int m0   = blockIdx.x * 128;
    const int n0   = blockIdx.y * 64;

    v8f acc[4];
#pragma unroll
    for (int j = 0; j < 4; ++j)
#pragma unroll
        for (int i = 0; i < 8; ++i) acc[j][i] = 0.0f;

    const int ar   = tid >> 1;          // 0..127 : A row within tile
    const int ak   = (tid & 1) * 32;    // 0/32   : k half-chunk
    const int bn   = tid >> 2;          // 0..63  : B row (=N col) within tile
    const int bq   = (tid & 3) * 16;    // k sub-chunk of 16 f16
    const int m16  = lane & 15, hh2 = lane >> 4;
    // Rows beyond M only feed output rows that are never stored: clamp, no zero.
    const int rowc = (m0 + ar < M) ? (m0 + ar) : (M - 1);
    const _Float16* brow = Bt + (size_t)(n0 + bn) * kpad + bq;

    // ---- prologue: stage tile k0=0 into buffer 0 ----
    float af[32];
    v8h   breg0, breg1;
    stage_a32(A, lda, rowc, kact, ak, af);
    breg0 = *(const v8h*)(brow);
    breg1 = *(const v8h*)(brow + 8);
    commit_a32(&As[0][ar * ASTR + ak], af);
    *(v8h*)(&Bs[0][bn * ASTR + bq])     = breg0;
    *(v8h*)(&Bs[0][bn * ASTR + bq + 8]) = breg1;
    __syncthreads();

    int buf = 0;
    for (int k0 = 0; k0 < kpad; k0 += 64) {
        const bool more = (k0 + 64) < kpad;
        if (more) {   // issue next tile's loads before compute (overlap WMMA)
            stage_a32(A, lda, rowc, kact, k0 + 64 + ak, af);
            breg0 = *(const v8h*)(brow + k0 + 64);
            breg1 = *(const v8h*)(brow + k0 + 64 + 8);
            if (k0 + 128 < kpad)   // global_prefetch_b8 of the tile after next
                __builtin_prefetch(A + (size_t)rowc * lda + k0 + 128 + ak, 0, 1);
        }

        const _Float16* Ab = As[buf];
        const _Float16* Bb = Bs[buf];

#pragma unroll
        for (int s = 0; s < 2; ++s) {   // two K=32 sub-steps per staged tile
            // A fragment (ISA 16-bit A layout: lane m holds row m, K by half)
            const _Float16* apf = &Ab[(w * 16 + m16) * ASTR + s * 32 + hh2 * 8];
            v8h alo = *(const v8h*)apf;
            v8h ahi = *(const v8h*)(apf + 16);
            v16h a;
#pragma unroll
            for (int i = 0; i < 8; ++i) { a[i] = alo[i]; a[i + 8] = ahi[i]; }

#pragma unroll
            for (int j = 0; j < 4; ++j) {
                const _Float16* bpf =
                    &Bb[(j * 16 + m16) * ASTR + s * 32 + hh2 * 8];
                v8h blo = *(const v8h*)bpf;
                v8h bhi = *(const v8h*)(bpf + 16);
                v16h b;
#pragma unroll
                for (int i = 0; i < 8; ++i) { b[i] = blo[i]; b[i + 8] = bhi[i]; }
                acc[j] = __builtin_amdgcn_wmma_f32_16x16x32_f16(
                    false, a, false, b, (short)0, acc[j], false, false);
            }
        }

        if (more) {   // commit next tile into the other buffer
            commit_a32(&As[buf ^ 1][ar * ASTR + ak], af);
            *(v8h*)(&Bs[buf ^ 1][bn * ASTR + bq])     = breg0;
            *(v8h*)(&Bs[buf ^ 1][bn * ASTR + bq + 8]) = breg1;
        }
        __syncthreads();
        buf ^= 1;
    }

    // ---- epilogue: C/D layout  M = 8*(lane>>4)+r , N = lane&15 ----
#pragma unroll
    for (int j = 0; j < 4; ++j) {
        const int col = n0 + j * 16 + m16;
        const float bv = bias ? bias[col] : 0.0f;
#pragma unroll
        for (int r = 0; r < 8; ++r) {
            const int row = m0 + w * 16 + hh2 * 8 + r;
            if (row < M) {
                float v = acc[j][r] + bv;
                if (mode == 1) {
                    v = fmaxf(v, 0.0f);
                } else if (mode == 2) {
                    v = g[col] * (v * BN_INV) + bb[col];
                    v = fmaxf(v, 0.0f);
                    if (res) v += res[(size_t)row * ldres + col];
                }
                C[(size_t)row * ldc + col] = v;
            }
        }
    }
}

// ---------------------------------------------------------------------------
// hv = h_slice + vn[batch] ; aggr = (1+eps[l]) * hv    (fused, float4)
// ---------------------------------------------------------------------------
__global__ void add_vn_kernel(const float* __restrict__ hsl, int ldh,
                              const float* __restrict__ vn,
                              const int* __restrict__ batch,
                              const float* __restrict__ eps, int l,
                              float* __restrict__ hv, float* __restrict__ aggr) {
    int t = blockIdx.x * blockDim.x + threadIdx.x;
    int nid = t >> 6, c = (t & 63) * 4;
    if (nid >= NN) return;
    float ep = 1.0f + eps[l];
    float4 hval = *(const float4*)(hsl + (size_t)nid * ldh + c);
    float4 vval = *(const float4*)(vn + (size_t)batch[nid] * HH + c);
    float4 o, ag;
    o.x = hval.x + vval.x; o.y = hval.y + vval.y;
    o.z = hval.z + vval.z; o.w = hval.w + vval.w;
    ag.x = ep * o.x; ag.y = ep * o.y; ag.z = ep * o.z; ag.w = ep * o.w;
    *(float4*)(hv + (size_t)nid * HH + c)   = o;
    *(float4*)(aggr + (size_t)nid * HH + c) = ag;
}

// ---------------------------------------------------------------------------
// Fused edge path: e = ea@We+be ; m = relu(hv[src]+e) ; aggr[dst] += m (atomic)
// We/be staged in LDS once per block; 8 threads/edge x 32 cols; float4 gathers.
// ---------------------------------------------------------------------------
__global__ __launch_bounds__(256)
void edge_msg_kernel(const float* __restrict__ ea,
                     const int* __restrict__ ei,
                     const float* __restrict__ We,
                     const float* __restrict__ be,
                     const float* __restrict__ hv,
                     float* __restrict__ aggr) {
    __shared__ float sW[13 * 256 + 256];
    for (int i = threadIdx.x; i < 13 * 256; i += 256) sW[i] = We[i];
    sW[13 * 256 + threadIdx.x] = be[threadIdx.x];
    __syncthreads();

    int t = blockIdx.x * 256 + threadIdx.x;
    int e = t >> 3;
    if (e >= NE) return;
    int c0  = (t & 7) * 32;
    int src = ei[e];
    int dst = ei[NE + e];
    float a[13];
#pragma unroll
    for (int k = 0; k < 13; ++k) a[k] = ea[(size_t)e * 13 + k];
    const float* hs = hv + (size_t)src * HH + c0;
    float* ag = aggr + (size_t)dst * HH + c0;
    const float* wb = sW + 13 * 256;
#pragma unroll
    for (int c4 = 0; c4 < 32; c4 += 4) {
        float4 h4 = *(const float4*)(hs + c4);
        float r[4] = {h4.x, h4.y, h4.z, h4.w};
#pragma unroll
        for (int u = 0; u < 4; ++u) {
            float acc = wb[c0 + c4 + u];
#pragma unroll
            for (int k = 0; k < 13; ++k)
                acc += a[k] * sW[k * 256 + c0 + c4 + u];
            r[u] = fmaxf(r[u] + acc, 0.0f);
        }
#pragma unroll
        for (int u = 0; u < 4; ++u) atomicAdd(&ag[c4 + u], r[u]);
    }
}

// ---------------------------------------------------------------------------
// segment bounds over sorted batch (atomicMin/Max -> deterministic)
// ---------------------------------------------------------------------------
__global__ void seg_init_kernel(int* __restrict__ s, int* __restrict__ e) {
    int g = blockIdx.x * blockDim.x + threadIdx.x;
    if (g < NG) { s[g] = 0x7fffffff; e[g] = 0; }
}
__global__ void seg_bounds_kernel(const int* __restrict__ batch,
                                  int* __restrict__ s, int* __restrict__ e) {
    int nid = blockIdx.x * blockDim.x + threadIdx.x;
    if (nid >= NN) return;
    int b = batch[nid];
    atomicMin(&s[b], nid);
    atomicMax(&e[b], nid + 1);
}

__global__ void vn_init_kernel(const float* __restrict__ emb0,
                               float* __restrict__ vn) {
    int t = blockIdx.x * blockDim.x + threadIdx.x;
    if (t < NG * HH) vn[t] = emb0[t & 255];
}

// vn_in[g] = sum_{nodes in g} h_slice + vn[g]   (block per graph, sorted batch)
__global__ void vn_agg_kernel(const float* __restrict__ hsl, int ldh,
                              const int* __restrict__ segs,
                              const int* __restrict__ sege,
                              const float* __restrict__ vn,
                              float* __restrict__ vn_in) {
    int g = blockIdx.x, c = threadIdx.x;
    int s = segs[g], e = sege[g];
    if (s >= e) { s = 0; e = 0; }
    float acc = 0.0f;
    for (int nid = s; nid < e; ++nid) acc += hsl[(size_t)nid * ldh + c];
    vn_in[(size_t)g * HH + c] = acc + vn[(size_t)g * HH + c];
}

// gate[n] = bg2 + dot(ghid[n,:], Wg2)  (wave per node, coalesced)
__global__ void gate_kernel(const float* __restrict__ ghid,
                            const float* __restrict__ Wg2,
                            const float* __restrict__ bg2,
                            float* __restrict__ gate) {
    int wid  = (blockIdx.x * blockDim.x + threadIdx.x) >> 5;
    int lane = threadIdx.x & 31;
    if (wid >= NN) return;
    float acc = 0.0f;
#pragma unroll
    for (int i = 0; i < 8; ++i) {
        int c = lane + i * 32;
        acc += ghid[(size_t)wid * HH + c] * Wg2[c];
    }
#pragma unroll
    for (int off = 16; off > 0; off >>= 1) acc += __shfl_xor(acc, off, 32);
    if (lane == 0) gate[wid] = acc + bg2[0];
}

// per-graph softmax pieces: ex = exp(gate - max), den = sum(ex)
__global__ void softmax_seg_kernel(const float* __restrict__ gate,
                                   const int* __restrict__ segs,
                                   const int* __restrict__ sege,
                                   float* __restrict__ ex,
                                   float* __restrict__ den) {
    __shared__ float red[64];
    int g = blockIdx.x, t = threadIdx.x;
    int s = segs[g], e = sege[g];
    if (s >= e) { s = 0; e = 0; }
    float mx = -__builtin_inff();
    for (int nid = s + t; nid < e; nid += 64) mx = fmaxf(mx, gate[nid]);
    red[t] = mx; __syncthreads();
    for (int off = 32; off > 0; off >>= 1) {
        if (t < off) red[t] = fmaxf(red[t], red[t + off]);
        __syncthreads();
    }
    mx = red[0]; __syncthreads();
    float sm = 0.0f;
    for (int nid = s + t; nid < e; nid += 64) {
        float v = __expf(gate[nid] - mx);
        ex[nid] = v;
        sm += v;
    }
    red[t] = sm; __syncthreads();
    for (int off = 32; off > 0; off >>= 1) {
        if (t < off) red[t] += red[t + off];
        __syncthreads();
    }
    if (t == 0) den[g] = red[0];
}

// pooled+concat: pc[g, 0:1536] = sum alpha*hjk ; pc[g, 1536:1546] = gfeat
__global__ void pool_kernel(const float* __restrict__ hjk,
                            const float* __restrict__ ex,
                            const float* __restrict__ den,
                            const int* __restrict__ segs,
                            const int* __restrict__ sege,
                            const float* __restrict__ gfeat,
                            float* __restrict__ pc) {
    int g = blockIdx.x, t = threadIdx.x;    // 256 threads
    int s = segs[g], e = sege[g];
    if (s >= e) { s = 0; e = 0; }
    float d   = den[g];
    float inv = (e > s && d != 0.0f) ? 1.0f / d : 0.0f;
#pragma unroll
    for (int chunk = 0; chunk < 6; ++chunk) {
        int col = chunk * 256 + t;
        float acc = 0.0f;
        for (int nid = s; nid < e; ++nid)
            acc += ex[nid] * hjk[(size_t)nid * JK + col];
        pc[(size_t)g * PCW + col] = acc * inv;
    }
    if (t < GF) pc[(size_t)g * PCW + JK + t] = gfeat[(size_t)g * GF + t];
}

// readout: out[g] = relu(pc@Wr1+br1) @ Wr2 + br2   (block per graph, 64 thr)
__global__ void readout_kernel(const float* __restrict__ pc,
                               const float* __restrict__ Wr1,
                               const float* __restrict__ br1,
                               const float* __restrict__ Wr2,
                               const float* __restrict__ br2,
                               float* __restrict__ out) {
    __shared__ float row[PCW];
    __shared__ float red[64];
    int g = blockIdx.x, t = threadIdx.x;    // 64 threads
    for (int i = t; i < PCW; i += 64) row[i] = pc[(size_t)g * PCW + i];
    __syncthreads();
    float acc = br1[t];
    for (int k = 0; k < PCW; ++k) acc += row[k] * Wr1[(size_t)k * 64 + t];
    red[t] = fmaxf(acc, 0.0f) * Wr2[t];
    __syncthreads();
    for (int off = 32; off > 0; off >>= 1) {
        if (t < off) red[t] += red[t + off];
        __syncthreads();
    }
    if (t == 0) out[g] = red[0] + br2[0];
}

// ---------------------------------------------------------------------------
extern "C" void kernel_launch(void* const* d_in, const int* in_sizes, int n_in,
                              void* d_out, int out_size, void* d_ws, size_t ws_size,
                              hipStream_t stream) {
    const float* x      = (const float*)d_in[0];
    const float* ea     = (const float*)d_in[1];
    const float* gfeat  = (const float*)d_in[2];
    const int*   ei     = (const int*)d_in[3];
    const int*   batch  = (const int*)d_in[4];
    const float* Wp     = (const float*)d_in[5];
    const float* bp     = (const float*)d_in[6];
    const float* vnemb0 = (const float*)d_in[7];
    const float* We     = (const float*)d_in[8];
    const float* be     = (const float*)d_in[9];
    const float* W1     = (const float*)d_in[10];
    const float* b1     = (const float*)d_in[11];
    const float* W2     = (const float*)d_in[12];
    const float* b2     = (const float*)d_in[13];
    const float* eps    = (const float*)d_in[14];
    const float* bn_g   = (const float*)d_in[15];
    const float* bn_b   = (const float*)d_in[16];
    const float* vn_W   = (const float*)d_in[17];
    const float* vn_b   = (const float*)d_in[18];
    const float* vn_g   = (const float*)d_in[19];
    const float* vn_bb  = (const float*)d_in[20];
    const float* Wg1    = (const float*)d_in[21];
    const float* bg1    = (const float*)d_in[22];
    const float* Wg2    = (const float*)d_in[23];
    const float* bg2    = (const float*)d_in[24];
    const float* Wr1    = (const float*)d_in[25];
    const float* br1    = (const float*)d_in[26];
    const float* Wr2    = (const float*)d_in[27];
    const float* br2    = (const float*)d_in[28];
    float* out = (float*)d_out;

    // ---- workspace carve-up (256B aligned) ----
    size_t off = 0;
    auto take = [&](size_t bytes) -> void* {
        void* p = (char*)d_ws + off;
        off = (off + bytes + 255) & ~(size_t)255;
        return p;
    };
    float* hjk   = (float*)take((size_t)NN * JK * 4);   // h_list concat
    float* hv    = (float*)take((size_t)NN * HH * 4);   // h + vn[batch]
    float* aggr  = (float*)take((size_t)NN * HH * 4);   // (1+eps)h + msgs
    float* t1    = (float*)take((size_t)NN * HH * 4);   // MLP hidden
    float* vn    = (float*)take((size_t)NG * HH * 4);
    float* vn_in = (float*)take((size_t)NG * HH * 4);
    float* gate  = (float*)take((size_t)NN * 4);
    float* ex    = (float*)take((size_t)NN * 4);
    int*   segs  = (int*)take((size_t)NG * 4);
    int*   sege  = (int*)take((size_t)NG * 4);
    float* den   = (float*)take((size_t)NG * 4);
    float* pc    = (float*)take((size_t)NG * PCW * 4);
    _Float16* Wpt  = (_Float16*)take((size_t)256 * 64 * 2);
    _Float16* W1t  = (_Float16*)take((size_t)LL * HH * HH * 2);
    _Float16* W2t  = (_Float16*)take((size_t)LL * HH * HH * 2);
    _Float16* vnWt = (_Float16*)take((size_t)LL * HH * HH * 2);
    _Float16* Wg1t = (_Float16*)take((size_t)HH * JK * 2);
    float* ghid = aggr;   // reuse after layer loop

    // ---- weight prep (f32 -> transposed f16) ----
    prep_wt_kernel<<<(256 * 64 + 255) / 256, 256, 0, stream>>>(Wp, 38, 256, 64, Wpt);
    for (int l = 0; l < LL; ++l) {
        prep_wt_kernel<<<256, 256, 0, stream>>>(W1 + (size_t)l * HH * HH, HH, HH, HH,
                                                W1t + (size_t)l * HH * HH);
        prep_wt_kernel<<<256, 256, 0, stream>>>(W2 + (size_t)l * HH * HH, HH, HH, HH,
                                                W2t + (size_t)l * HH * HH);
        prep_wt_kernel<<<256, 256, 0, stream>>>(vn_W + (size_t)l * HH * HH, HH, HH, HH,
                                                vnWt + (size_t)l * HH * HH);
    }
    prep_wt_kernel<<<1536, 256, 0, stream>>>(Wg1, JK, HH, JK, Wg1t);

    // ---- segment bounds (batch is sorted) + vn init ----
    seg_init_kernel<<<(NG + 255) / 256, 256, 0, stream>>>(segs, sege);
    seg_bounds_kernel<<<(NN + 255) / 256, 256, 0, stream>>>(batch, segs, sege);
    vn_init_kernel<<<(NG * HH + 255) / 256, 256, 0, stream>>>(vnemb0, vn);

    // ---- node projection: hjk[:,0:256] = x @ Wp + bp ----
    dim3 gN((NN + 127) / 128, HH / 64);
    wmma_gemm_kernel<<<gN, 256, 0, stream>>>(x, 38, Wpt, 64, hjk, JK, bp,
                                             nullptr, nullptr, nullptr, 0,
                                             NN, 38, 0);

    // ---- GINE layers ----
    for (int l = 0; l < LL; ++l) {
        add_vn_kernel<<<(NN * 64 + 255) / 256, 256, 0, stream>>>(
            hjk + (size_t)l * HH, JK, vn, batch, eps, l, hv, aggr);
        edge_msg_kernel<<<(NE * 8 + 255) / 256, 256, 0, stream>>>(
            ea, ei, We + (size_t)l * 13 * HH, be + (size_t)l * HH, hv, aggr);
        // t1 = relu(aggr @ W1 + b1)
        wmma_gemm_kernel<<<gN, 256, 0, stream>>>(
            aggr, HH, W1t + (size_t)l * HH * HH, HH, t1, HH, b1 + (size_t)l * HH,
            nullptr, nullptr, nullptr, 0, NN, HH, 1);
        // hjk[:, (l+1)*256 : ] = relu(BN(t1 @ W2 + b2)) + hv
        wmma_gemm_kernel<<<gN, 256, 0, stream>>>(
            t1, HH, W2t + (size_t)l * HH * HH, HH, hjk + (size_t)(l + 1) * HH, JK,
            b2 + (size_t)l * HH, bn_g + (size_t)l * HH, bn_b + (size_t)l * HH,
            hv, HH, NN, HH, 2);
        // virtual node: vn_in = segsum(h_new) + vn ; vn = relu(BN(vn_in@vnW+b))
        vn_agg_kernel<<<NG, 256, 0, stream>>>(hjk + (size_t)(l + 1) * HH, JK,
                                              segs, sege, vn, vn_in);
        dim3 gV((NG + 127) / 128, HH / 64);
        wmma_gemm_kernel<<<gV, 256, 0, stream>>>(
            vn_in, HH, vnWt + (size_t)l * HH * HH, HH, vn, HH,
            vn_b + (size_t)l * HH, vn_g + (size_t)l * HH, vn_bb + (size_t)l * HH,
            nullptr, 0, NG, HH, 2);
    }

    // ---- attention gate: ghid = relu(hjk @ Wg1 + bg1); gate = ghid@Wg2+bg2 ----
    wmma_gemm_kernel<<<gN, 256, 0, stream>>>(hjk, JK, Wg1t, JK, ghid, HH, bg1,
                                             nullptr, nullptr, nullptr, 0,
                                             NN, JK, 1);
    gate_kernel<<<(NN * 32 + 255) / 256, 256, 0, stream>>>(ghid, Wg2, bg2, gate);

    // ---- per-graph softmax + pooling + concat global features ----
    softmax_seg_kernel<<<NG, 64, 0, stream>>>(gate, segs, sege, ex, den);
    pool_kernel<<<NG, 256, 0, stream>>>(hjk, ex, den, segs, sege, gfeat, pc);

    // ---- readout MLP -> out[G,1] ----
    readout_kernel<<<NG, 64, 0, stream>>>(pc, Wr1, br1, Wr2, br2, out);
}
